// W4A8B8O8LinearParallel_40243843564266
// MI455X (gfx1250) — 8-bit matrix operand order repaired and checked by reading, compiled, not run on hardware
//
#include <hip/hip_runtime.h>
#include <stdint.h>

typedef __attribute__((ext_vector_type(8))) int v8i;

#define M_DIM 4096
#define K_DIM 4096
#define N_DIM 4096
#define GRP   128

#define BM 64
#define BN 128
#define BK 128

#ifndef __has_builtin
#define __has_builtin(x) 0
#endif

#if __has_builtin(__builtin_amdgcn_global_load_async_to_lds_b128)
#define USE_ASYNC_LDS 1
#else
#define USE_ASYNC_LDS 0
#endif
#if __has_builtin(__builtin_amdgcn_s_wait_asynccnt)
#define HAVE_WAIT_ASYNC 1
#else
#define HAVE_WAIT_ASYNC 0
#endif

typedef int vi4_t __attribute__((vector_size(16)));
typedef vi4_t __attribute__((address_space(1)))* async_gptr_t;
typedef vi4_t __attribute__((address_space(3)))* async_lptr_t;

__device__ __forceinline__ void copy_b128_to_lds(const int8_t* g, int8_t* l) {
#if USE_ASYNC_LDS
  __builtin_amdgcn_global_load_async_to_lds_b128(
      (async_gptr_t)(uintptr_t)g, (async_lptr_t)(uintptr_t)l, 0, 0);
#else
  *(int4*)l = *(const int4*)g;
#endif
}

__device__ __forceinline__ void wait_async_copies() {
#if USE_ASYNC_LDS
#if HAVE_WAIT_ASYNC
  __builtin_amdgcn_s_wait_asynccnt(0);
#else
  asm volatile("s_wait_asynccnt 0" ::: "memory");
#endif
#endif
}

__global__ __launch_bounds__(256) void pack_x_kernel(const int* __restrict__ x,
                                                     int8_t* __restrict__ x8) {
  int idx = blockIdx.x * 1024 + threadIdx.x * 4;
  int4 v = *(const int4*)(x + idx);
  uint32_t p = (uint32_t)(v.x & 255) | ((uint32_t)(v.y & 255) << 8) |
               ((uint32_t)(v.z & 255) << 16) | ((uint32_t)(v.w & 255) << 24);
  *(uint32_t*)(x8 + idx) = p;
}

__global__ __launch_bounds__(256) void pack_w_kernel(const int* __restrict__ qw,
                                                     int8_t* __restrict__ w8) {
  __shared__ int8_t tile[64][64 + 4];
  const int bk = blockIdx.y * 64;
  const int bn = blockIdx.x * 64;
#pragma unroll
  for (int i = 0; i < 16; ++i) {
    int e = threadIdx.x + i * 256;
    int r = e >> 6, c = e & 63;
    tile[c][r] = (int8_t)qw[(size_t)(bk + r) * N_DIM + bn + c];
  }
  __syncthreads();
#pragma unroll
  for (int i = 0; i < 16; ++i) {
    int e = threadIdx.x + i * 256;
    int r = e >> 6, c = e & 63;
    w8[(size_t)(bn + r) * K_DIM + bk + c] = tile[r][c];
  }
}

__global__ __launch_bounds__(256) void w4a8_wmma_kernel(
    const int8_t* __restrict__ A8,
    const int8_t* __restrict__ B8,
    const float* __restrict__ in_scale,
    const float* __restrict__ s_channel,
    const _Float16* __restrict__ s_group,
    float* __restrict__ Y)
{
  __shared__ __align__(16) int8_t sA[2][BM][BK];
  __shared__ __align__(16) int8_t sB[2][BN][BK];

  const int tid  = threadIdx.x;
  const int lane = tid & 31;
  const int wave = tid >> 5;
  const int wm   = (wave >> 2) * 32;
  const int wn   = (wave & 3) * 32;
  const int hl   = lane >> 4;
  const int l16  = lane & 15;

  const int m0 = blockIdx.y * BM;
  const int n0 = blockIdx.x * BN;

  const int a_r = (tid * 16) >> 7, a_c = (tid * 16) & 127;

  float facc[2][2][8];
#pragma unroll
  for (int a = 0; a < 2; ++a)
#pragma unroll
    for (int b = 0; b < 2; ++b)
#pragma unroll
      for (int i = 0; i < 8; ++i) facc[a][b][i] = 0.0f;

  const float isc = in_scale[0];

  auto stage = [&](int g, int buf) {
    const int k0 = g * BK;
    copy_b128_to_lds(&A8[(size_t)(m0 + a_r) * K_DIM + k0 + a_c],
                     &sA[buf][a_r][a_c]);
    copy_b128_to_lds(&A8[(size_t)(m0 + a_r + 32) * K_DIM + k0 + a_c],
                     &sA[buf][a_r + 32][a_c]);
#pragma unroll
    for (int i = 0; i < 4; ++i) {
      const int off = tid * 16 + i * 4096;
      const int r = off >> 7, c = off & 127;
      copy_b128_to_lds(&B8[(size_t)(n0 + r) * K_DIM + k0 + c],
                       &sB[buf][r][c]);
    }
  };

  stage(0, 0);
  wait_async_copies();
  __syncthreads();

  int buf = 0;
  const int NG = K_DIM / BK;
  for (int g = 0; g < NG; ++g) {
    if (g + 1 < NG) stage(g + 1, buf ^ 1);

    v8i afrag[2][2];
#pragma unroll
    for (int mt = 0; mt < 2; ++mt) {
      const int8_t* row = &sA[buf][wm + mt * 16 + l16][0];
#pragma unroll
      for (int kk = 0; kk < 2; ++kk) {
        union { v8i v; int64_t d[4]; } u;
        const int base = kk * 64 + hl * 16;
#pragma unroll
        for (int p = 0; p < 4; ++p)
          u.d[p] = *(const int64_t*)(row + base + (p & 1) * 8 + (p >> 1) * 32);
        afrag[mt][kk] = u.v;
      }
    }
    v8i bfrag[2][2];
#pragma unroll
    for (int nt = 0; nt < 2; ++nt) {
      const int8_t* col = &sB[buf][wn + nt * 16 + l16][0];
#pragma unroll
      for (int kk = 0; kk < 2; ++kk) {
        union { v8i v; int4 q[2]; } u;
        u.q[0] = *(const int4*)(col + kk * 64 + hl * 16);
        u.q[1] = *(const int4*)(col + kk * 64 + 32 + hl * 16);
        bfrag[nt][kk] = u.v;
      }
    }

    float sg[2];
    sg[0] = (float)s_group[(size_t)g * N_DIM + n0 + wn + l16];
    sg[1] = (float)s_group[(size_t)g * N_DIM + n0 + wn + 16 + l16];

#pragma unroll
    for (int mt = 0; mt < 2; ++mt) {
#pragma unroll
      for (int nt = 0; nt < 2; ++nt) {
        v8i iacc = {0, 0, 0, 0, 0, 0, 0, 0};
        iacc = __builtin_amdgcn_wmma_i32_16x16x64_iu8(
            true, afrag[mt][0], true, bfrag[nt][0], iacc, false, false);
        iacc = __builtin_amdgcn_wmma_i32_16x16x64_iu8(
            true, afrag[mt][1], true, bfrag[nt][1], iacc, false, false);
#pragma unroll
        for (int i = 0; i < 8; ++i)
          facc[mt][nt][i] += (float)iacc[i] * sg[nt];
      }
    }

    wait_async_copies();
    __syncthreads();
    buf ^= 1;
  }

#pragma unroll
  for (int mt = 0; mt < 2; ++mt) {
#pragma unroll
    for (int nt = 0; nt < 2; ++nt) {
      const int col = n0 + wn + nt * 16 + l16;
      const float cs = s_channel[col] * isc;
#pragma unroll
      for (int i = 0; i < 8; ++i) {
        const int row = m0 + wm + mt * 16 + hl * 8 + i;
        Y[(size_t)row * N_DIM + col] = facc[mt][nt][i] * cs;
      }
    }
  }
}

extern "C" void kernel_launch(void* const* d_in, const int* in_sizes, int n_in,
                              void* d_out, int out_size, void* d_ws, size_t ws_size,
                              hipStream_t stream) {
  const int*      x      = (const int*)d_in[0];
  const int*      qw     = (const int*)d_in[1];
  const float*    iscale = (const float*)d_in[2];
  const float*    schan  = (const float*)d_in[3];
  const _Float16* sgroup = (const _Float16*)d_in[4];
  float*          Y      = (float*)d_out;

  int8_t* x8 = (int8_t*)d_ws;
  int8_t* w8 = x8 + (size_t)M_DIM * K_DIM;

  pack_x_kernel<<<(M_DIM * K_DIM) / 1024, 256, 0, stream>>>(x, x8);

  dim3 gw(N_DIM / 64, K_DIM / 64);
  pack_w_kernel<<<gw, 256, 0, stream>>>(qw, w8);

  dim3 gg(N_DIM / BN, M_DIM / BM);
  w4a8_wmma_kernel<<<gg, 256, 0, stream>>>(x8, w8, iscale, schan, sgroup, Y);
}
